// YOLOV5TorchObjectDetector_39075612459797
// MI455X (gfx1250) — compile-verified, matrix operand order checked by reading, not run on hardware
//
#include <hip/hip_runtime.h>
#include <cstdint>

#define BATCH    16
#define NPRED    25200
#define NCLS     80
#define PROW     85
#define KPRE     1024
#define MAXDET   300
#define OUTROW   89
#define CONF_T   0.4f
#define IOU_T    0.45f
#define MAX_WH_F 4096.0f
#define NBINS    4096

typedef __attribute__((ext_vector_type(16))) _Float16 v16h;
typedef __attribute__((ext_vector_type(8)))  float    v8f;
typedef int v4i_vs __attribute__((vector_size(16)));   // matches async-LDS builtin param

__device__ __forceinline__ float sigmoidf(float x) {
  return 1.0f / (1.0f + __expf(-x));
}

__device__ __forceinline__ unsigned long long pack_key(float s, unsigned idx) {
  return ((unsigned long long)__float_as_uint(s) << 32) | (unsigned long long)idx;
}

// Descending bitonic sort of KPRE 64-bit keys in LDS; blockDim.x == KPRE.
__device__ __forceinline__ void bitonic_desc(unsigned long long* keys, int tid) {
  for (int k = 2; k <= KPRE; k <<= 1) {
    for (int j = k >> 1; j > 0; j >>= 1) {
      int ixj = tid ^ j;
      if (ixj > tid) {
        unsigned long long a = keys[tid];
        unsigned long long c = keys[ixj];
        bool up = (tid & k) == 0;
        bool sw = up ? (a < c) : (a > c);
        if (sw) { keys[tid] = c; keys[ixj] = a; }
      }
      __syncthreads();
    }
  }
}

// ---------------------------------------------------------------------------
// Kernel 1: per-prediction score / class (bandwidth bound, ~137 MB total read)
// ---------------------------------------------------------------------------
__global__ void k_scores(const float* __restrict__ pred,
                         float* __restrict__ score, int* __restrict__ cls) {
  int t = blockIdx.x * blockDim.x + threadIdx.x;
  if (t >= BATCH * NPRED) return;
  const float* p = pred + (size_t)t * PROW;
  float obj = p[4];
  float best = -1.0f; int bj = 0;
  #pragma unroll 4
  for (int c = 0; c < NCLS; ++c) {
    float v = p[5 + c];
    if (v > best) { best = v; bj = c; }
  }
  float conf = best * obj;
  bool valid = (obj > CONF_T) && (conf > CONF_T);
  score[t] = valid ? conf : 0.0f;
  cls[t] = bj;
}

// ---------------------------------------------------------------------------
// Kernel 2: per-batch top-1024 via LDS histogram threshold + bitonic sort
// ---------------------------------------------------------------------------
__global__ void k_topk(const float* __restrict__ score,
                       int* __restrict__ topi, float* __restrict__ tops) {
  __shared__ unsigned bins[NBINS];
  __shared__ unsigned long long keys[KPRE];
  __shared__ int s_thr;
  __shared__ unsigned s_above, s_c0, s_c1;

  const int b = blockIdx.x;
  const int tid = threadIdx.x;
  const float* sc = score + (size_t)b * NPRED;

  for (int i = tid; i < NBINS; i += KPRE) bins[i] = 0u;
  if (tid == 0) { s_c0 = 0u; s_c1 = 0u; }
  keys[tid] = 0ull;
  __syncthreads();

  for (int i = tid; i < NPRED; i += KPRE) {
    float s = sc[i];
    if (s > 0.0f) {
      int bb = (int)(s * (float)NBINS);
      bb = bb > NBINS - 1 ? NBINS - 1 : bb;
      atomicAdd(&bins[bb], 1u);
    }
  }
  __syncthreads();

  if (tid == 0) {
    unsigned cum = 0; int thr = -1; unsigned above = 0;
    for (int bb = NBINS - 1; bb >= 0; --bb) {
      unsigned c = bins[bb];
      if (cum + c >= KPRE) { thr = bb; above = cum; break; }
      cum += c;
    }
    if (thr < 0) { thr = -1; above = 0; }   // fewer than KPRE positives: take all
    s_thr = thr; s_above = above;
  }
  __syncthreads();

  const int thr = s_thr;
  const unsigned above = s_above;
  for (int i = tid; i < NPRED; i += KPRE) {
    float s = sc[i];
    if (s <= 0.0f) continue;
    int bb = (int)(s * (float)NBINS);
    bb = bb > NBINS - 1 ? NBINS - 1 : bb;
    if (bb > thr) {
      unsigned pos = atomicAdd(&s_c0, 1u);
      if (pos < KPRE) keys[pos] = pack_key(s, (unsigned)i);
    } else if (bb == thr) {
      unsigned pos = above + atomicAdd(&s_c1, 1u);
      if (pos < KPRE) keys[pos] = pack_key(s, (unsigned)i);
    }
  }
  __syncthreads();

  bitonic_desc(keys, tid);

  unsigned long long kk = keys[tid];
  tops[(size_t)b * KPRE + tid] = __uint_as_float((unsigned)(kk >> 32));
  topi[(size_t)b * KPRE + tid] = (int)(kk & 0xFFFFFFFFu);
}

// ---------------------------------------------------------------------------
// Kernel 3: per-batch NMS. Full 1024x1024 suppression bitmask in LDS (128 KB,
// CDNA5 320KB WGP LDS). Union denominator (area_i + area_j) computed per
// 16x16 tile with v_wmma_f32_16x16x32_f16 as a rank-2 outer product.
// IoU threshold test is division-free: inter/(U) > T  <=>  inter > T*U, U>0.
// Dynamic LDS layout:
//   [0,      16384)  float4 s_box[1024]
//   [16384,  20480)  float  s_tops[1024]
//   [20480,  20608)  uint   s_keep[32]
//   [20608, 151680)  uint   s_sup[1024][32]
//   [151680,153728)  _Float16 s_area[1024]
// ---------------------------------------------------------------------------
#define NMS_SHMEM 153728

__global__ void k_nms(const float* __restrict__ pred,
                      const int* __restrict__ cls,
                      const int* __restrict__ topi,
                      const float* __restrict__ tops,
                      float* __restrict__ finals) {
  extern __shared__ __align__(16) char smem[];
  float4*    s_box  = (float4*)(smem);
  float*     s_tops = (float*)(smem + 16384);
  unsigned*  s_keep = (unsigned*)(smem + 20480);
  unsigned*  s_sup  = (unsigned*)(smem + 20608);
  _Float16*  s_area = (_Float16*)(smem + 151680);

  const int b   = blockIdx.x;
  const int tid = threadIdx.x;
  const int lane = tid & 31;
  const int wv   = tid >> 5;

  const int*   topi_b = topi + (size_t)b * KPRE;
  const float* tops_b = tops + (size_t)b * KPRE;

  // Stage top_s into LDS via CDNA5 async global->LDS DMA when available.
#if __has_builtin(__builtin_amdgcn_global_load_async_to_lds_b128)
  if (tid < 256) {
    __builtin_amdgcn_global_load_async_to_lds_b128(
        (__attribute__((address_space(1))) v4i_vs*)(uintptr_t)(tops_b + tid * 4),
        (__attribute__((address_space(3))) v4i_vs*)(s_tops + tid * 4),
        0, 0);
  }
#if __has_builtin(__builtin_amdgcn_s_wait_asynccnt)
  __builtin_amdgcn_s_wait_asynccnt(0);
#else
  asm volatile("s_wait_asynccnt 0" ::: "memory");
#endif
#else
  s_tops[tid] = tops_b[tid];
#endif

  // Class-offset boxes + fp16 areas.
  {
    int gi = topi_b[tid];
    const float* p = pred + ((size_t)b * NPRED + (size_t)gi) * PROW;
    float cx = p[0], cy = p[1], w = p[2], h = p[3];
    float off = (float)cls[(size_t)b * NPRED + gi] * MAX_WH_F;
    s_box[tid] = make_float4(cx - 0.5f * w + off, cy - 0.5f * h + off,
                             cx + 0.5f * w + off, cy + 0.5f * h + off);
    s_area[tid] = (_Float16)(w * h);
  }
  #pragma unroll
  for (int wdx = 0; wdx < 32; ++wdx) s_sup[tid * 32 + wdx] = 0u;
  __syncthreads();

  // Build suppression bitmask: 64x64 grid of 16x16 tiles, 32 waves, 128 tiles
  // per wave. One WMMA per tile yields D[M][N] = area_row[M] + area_col[N].
  const bool lo = lane < 16;
  for (int tile = wv; tile < 64 * 64; tile += 32) {
    int tr = tile >> 6;
    int tc = tile & 63;
    _Float16 ar = s_area[tr * 16 + (lane & 15)];
    _Float16 ac = s_area[tc * 16 + (lane & 15)];
    v16h A = {};
    v16h Bv = {};
    A[0]  = lo ? ar : (_Float16)0.0f;              // K=0 : area of row M=lane
    A[1]  = lo ? (_Float16)1.0f : (_Float16)0.0f;  // K=1 : one
    Bv[0] = lo ? (_Float16)1.0f : (_Float16)0.0f;  // K=0 : one
    Bv[1] = lo ? ac : (_Float16)0.0f;              // K=1 : area of col N=lane
    v8f acc = {};
    acc = __builtin_amdgcn_wmma_f32_16x16x32_f16(
        false, A, false, Bv, (short)0, acc, false, false);

    int cN = tc * 16 + (lane & 15);
    float4 bc = s_box[cN];
    #pragma unroll
    for (int k = 0; k < 8; ++k) {
      int rM = tr * 16 + (lo ? k : k + 8);
      float4 br = s_box[rM];
      float lx = fmaxf(br.x, bc.x), ly = fmaxf(br.y, bc.y);
      float rx = fminf(br.z, bc.z), ry = fminf(br.w, bc.w);
      float inter = fmaxf(rx - lx, 0.0f) * fmaxf(ry - ly, 0.0f);
      // inter/(sum - inter + eps) > T  <=>  inter > T*(sum - inter + eps)
      float uni = acc[k] - inter + 1e-7f;          // always > 0
      if (inter > IOU_T * uni && rM != cN)
        atomicOr(&s_sup[rM * 32 + (cN >> 5)], 1u << (cN & 31));
    }
  }
  __syncthreads();

  // keep-mask init from valid_k = top_s > 0
  if (tid < 32) {
    unsigned wbits = 0u;
    for (int i = 0; i < 32; ++i)
      if (s_tops[tid * 32 + i] > 0.0f) wbits |= 1u << i;
    s_keep[tid] = wbits;
  }
  __syncthreads();

  // Greedy sequential suppression scan: wave 0, 32 lanes own 32 keep words.
  if (tid < 32) {
    volatile unsigned* vk = s_keep;
    for (int i = 0; i < KPRE; ++i) {
      unsigned kw = vk[i >> 5];
      if ((kw >> (i & 31)) & 1u) {
        unsigned s = s_sup[i * 32 + tid];
        int iw = i >> 5, ib = i & 31;
        if (tid < iw)       s = 0u;
        else if (tid == iw) s &= ~((2u << ib) - 1u);   // keep only bits j > i
        vk[tid] &= ~s;
      }
    }
  }
  __syncthreads();

  finals[(size_t)b * KPRE + tid] =
      ((s_keep[tid >> 5] >> (tid & 31)) & 1u) ? s_tops[tid] : 0.0f;
}

// ---------------------------------------------------------------------------
// Kernel 4: top-300 of kept scores, gather, sigmoid only for the 300 winners.
// ---------------------------------------------------------------------------
__global__ void k_final(const float* __restrict__ pred,
                        const float* __restrict__ clogit,
                        const float* __restrict__ logits,
                        const float* __restrict__ head,
                        const float* __restrict__ finals,
                        const int* __restrict__ topi,
                        const int* __restrict__ cls,
                        float* __restrict__ out) {
  __shared__ unsigned long long keys[KPRE];
  const int b = blockIdx.x;
  const int tid = threadIdx.x;

  keys[tid] = pack_key(finals[(size_t)b * KPRE + tid], (unsigned)tid);
  __syncthreads();
  bitonic_desc(keys, tid);

  if (tid < MAXDET) {
    unsigned long long kk = keys[tid];
    float sc = __uint_as_float((unsigned)(kk >> 32));
    int slot = (int)(kk & 0xFFFFFFFFu);
    float m = sc > 0.0f ? 1.0f : 0.0f;
    int gi = topi[(size_t)b * KPRE + slot];
    const float* p = pred + ((size_t)b * NPRED + (size_t)gi) * PROW;
    float cx = p[0], cy = p[1], w = p[2], h = p[3];
    float* o = out + ((size_t)b * MAXDET + tid) * OUTROW;
    o[0] = (cx - 0.5f * w) * m;
    o[1] = (cy - 0.5f * h) * m;
    o[2] = (cx + 0.5f * w) * m;
    o[3] = (cy + 0.5f * h) * m;
    o[4] = sc * m;
    o[5] = (float)cls[(size_t)b * NPRED + gi] * m;
    float obj_sig = sigmoidf(clogit[((size_t)b * NPRED + gi) * 5 + 4]);
    const float* lg = logits + ((size_t)b * NPRED + (size_t)gi) * NCLS;
    for (int c = 0; c < NCLS; ++c)
      o[6 + c] = sigmoidf(lg[c]) * obj_sig * m;
    o[86] = obj_sig * m;
    o[87] = head[(size_t)b * NPRED + gi] * m;
    o[88] = m;
  }
}

// ---------------------------------------------------------------------------
extern "C" void kernel_launch(void* const* d_in, const int* in_sizes, int n_in,
                              void* d_out, int out_size, void* d_ws, size_t ws_size,
                              hipStream_t stream) {
  (void)in_sizes; (void)n_in; (void)out_size; (void)ws_size;
  const float* pred   = (const float*)d_in[0];
  const float* clogit = (const float*)d_in[1];
  const float* logits = (const float*)d_in[2];
  const float* head   = (const float*)d_in[3];
  float* out = (float*)d_out;

  char* ws = (char*)d_ws;
  const size_t BN = (size_t)BATCH * NPRED;
  const size_t BK = (size_t)BATCH * KPRE;
  float* ws_score = (float*)(ws);
  int*   ws_cls   = (int*)(ws + BN * 4);
  int*   ws_topi  = (int*)(ws + 2 * BN * 4);
  float* ws_tops  = (float*)(ws + 2 * BN * 4 + BK * 4);
  float* ws_final = (float*)(ws + 2 * BN * 4 + 2 * BK * 4);

  const int total = BATCH * NPRED;
  k_scores<<<dim3((total + 255) / 256), dim3(256), 0, stream>>>(pred, ws_score, ws_cls);
  k_topk<<<dim3(BATCH), dim3(KPRE), 0, stream>>>(ws_score, ws_topi, ws_tops);
  k_nms<<<dim3(BATCH), dim3(KPRE), NMS_SHMEM, stream>>>(pred, ws_cls, ws_topi,
                                                        ws_tops, ws_final);
  k_final<<<dim3(BATCH), dim3(KPRE), 0, stream>>>(pred, clogit, logits, head,
                                                  ws_final, ws_topi, ws_cls, out);
}
